// RNN_52132313039289
// MI455X (gfx1250) — matrix operand order repaired and checked by reading, not compiled, not run on hardware
//
#include <hip/hip_runtime.h>

// ---------------------------------------------------------------------------
// RNN pipeline for MI455X (gfx1250, wave32, WMMA)
//   logits = ((emb[X] Wxh^T + b) -> recurrence(Whh) -> Wyh^T -> Wlo^T)
// Big GEMMs: bf16 in / f32 accumulate via v_wmma_f32_16x16x32_bf16,
//   double-buffered LDS staging via GLOBAL_LOAD_ASYNC_TO_LDS_B128 with
//   split-counter pipelining (s_wait_asynccnt 3 retires prev tile while the
//   next streams).
// Recurrence: fp32 via v_wmma_f32_16x16x4_f32, h-state in LDS.
// ---------------------------------------------------------------------------

#define USE_ASYNC_LDS 1   // set 0 to fall back to global_load + ds_store

typedef __bf16 bf16;
typedef __attribute__((ext_vector_type(16))) bf16  v16bf;
typedef __attribute__((ext_vector_type(8)))  bf16  v8bf;
typedef __attribute__((ext_vector_type(8)))  float v8f;
typedef __attribute__((ext_vector_type(2)))  float v2f;

union FragBF { v16bf v; v8bf h[2]; };

static constexpr int cB = 16, cT = 512, cV = 8000, cE = 512, cH = 1024;

// --- gfx1250 async-to-LDS helpers -----------------------------------------
__device__ __forceinline__ unsigned lds_offset(const void* p) {
  // generic -> addrspace(3) cast; ptrtoint of an LDS pointer gives the 32-bit
  // byte offset from LDS_BASE (the async instruction's VDST operand).
  return (unsigned)(unsigned long long)
      (__attribute__((address_space(3))) const void*)p;
}
__device__ __forceinline__ void async_load_b128(unsigned loff, const void* g) {
  asm volatile("global_load_async_to_lds_b128 %0, %1, off"
               :: "v"(loff), "v"(g) : "memory");
}
template <int N>
__device__ __forceinline__ void wait_asynccnt() {
  asm volatile("s_wait_asynccnt %0" :: "i"(N) : "memory");
}

// ---------------------------------------------------------------------------
// Elementwise fp32 -> bf16 cast (weight preconversion)
// ---------------------------------------------------------------------------
__global__ void cast_f32_bf16_kernel(const float* __restrict__ src,
                                     bf16* __restrict__ dst, int n) {
  int i = blockIdx.x * blockDim.x + threadIdx.x;
  if (i < n) dst[i] = (bf16)src[i];
}

// ---------------------------------------------------------------------------
// Embedding gather fused with bf16 cast: Xe[i,e] = emb[X[i], e]
// ---------------------------------------------------------------------------
__global__ void embed_gather_kernel(const int* __restrict__ X,
                                    const float* __restrict__ emb,
                                    bf16* __restrict__ Xe, int n, int E) {
  int i = blockIdx.x * blockDim.x + threadIdx.x;
  if (i < n) {
    int tok = i / E;
    int e   = i - tok * E;
    Xe[i] = (bf16)emb[(size_t)X[tok] * E + e];
  }
}

// ---------------------------------------------------------------------------
// C[M,N] = A[M,K](bf16) * W[N,K]^T(bf16) + bias[N]
// Block tile 128x64, BK=32, 8 waves, each wave 2x2 WMMA 16x16 subtiles.
// Double-buffered async-to-LDS pipeline. bias must be non-null.
// Requires: M%128==0, N%64==0, K%32==0 (true for all three call sites).
// ---------------------------------------------------------------------------
#define GM_BM 128
#define GM_BN 64
#define GM_BK 32
#define GM_LDA 40   // padded bf16 row stride: 80 bytes, 16B-aligned rows
#define GM_LDB 40

template <bool HAS_C, bool HAS_CB>
__global__ __launch_bounds__(256) void gemm_bf16_kernel(
    const bf16* __restrict__ A, const bf16* __restrict__ W,
    const float* __restrict__ bias,
    float* __restrict__ C, bf16* __restrict__ Cb,
    int M, int N, int K)
{
  __shared__ __align__(16) bf16 As[2][GM_BM * GM_LDA];  // 2 x 10 KB
  __shared__ __align__(16) bf16 Bs[2][GM_BN * GM_LDB];  // 2 x  5 KB

  const int tid  = threadIdx.x;
  const int lane = tid & 31;
  const int wave = tid >> 5;      // 0..7
  const int wm   = wave & 3;      // M group: 32*wm
  const int wn   = wave >> 2;     // N group: 32*wn
  const int m0   = blockIdx.y * GM_BM;
  const int n0   = blockIdx.x * GM_BN;
  const int g    = lane >> 4;     // lane half (0/1)
  const int mn   = lane & 15;     // row/col within 16x16 tile

  // Per-thread staging map (full K coverage):
  //   A tile 128x32 bf16 = 8 KB -> 32 B/thread: row ra, k in [ha*16, ha*16+16)
  //   B tile  64x32 bf16 = 4 KB -> 16 B/thread: row rb, k in [qb*8,  qb*8+8)
  const int ra = tid >> 1, ha = tid & 1;
  const int rb = tid >> 2, qb = tid & 3;
  unsigned As_l[2], Bs_l[2];
  As_l[0] = lds_offset(&As[0][ra * GM_LDA + ha * 16]);
  As_l[1] = lds_offset(&As[1][ra * GM_LDA + ha * 16]);
  Bs_l[0] = lds_offset(&Bs[0][rb * GM_LDB + qb * 8]);
  Bs_l[1] = lds_offset(&Bs[1][rb * GM_LDB + qb * 8]);
  const bf16* gA = A + (size_t)(m0 + ra) * K + ha * 16;  // advance by kk
  const bf16* gB = W + (size_t)(n0 + rb) * K + qb * 8;

  auto stage = [&](int buf, int kk) {
#if USE_ASYNC_LDS
    async_load_b128(As_l[buf],      gA + kk);
    async_load_b128(As_l[buf] + 16, gA + kk + 8);
    async_load_b128(Bs_l[buf],      gB + kk);
#else
    uint4 a0 = *(const uint4*)(gA + kk);
    uint4 a1 = *(const uint4*)(gA + kk + 8);
    uint4 b0 = *(const uint4*)(gB + kk);
    *(uint4*)(&As[buf][ra * GM_LDA + ha * 16])     = a0;
    *(uint4*)(&As[buf][ra * GM_LDA + ha * 16 + 8]) = a1;
    *(uint4*)(&Bs[buf][rb * GM_LDB + qb * 8])      = b0;
#endif
  };

  v8f acc[2][2];
  for (int i = 0; i < 2; ++i)
    for (int j = 0; j < 2; ++j)
      acc[i][j] = (v8f)0.f;

  stage(0, 0);                       // prologue: tile 0 -> buffer 0
  int p = 0;
  for (int kk = 0; kk < K; kk += GM_BK) {
    const bool has_next = (kk + GM_BK) < K;
    if (has_next) {
      stage(p ^ 1, kk + GM_BK);      // prefetch next tile into other buffer
      wait_asynccnt<3>();            // retire previous tile's 3 async ops
    } else {
      wait_asynccnt<0>();
    }
    __syncthreads();                 // buffer p fully populated (all waves)

    // Fragments per documented 16-bit A/B VGPR layouts:
    //  A lane(l): m=l%16; elems 0-7 -> K=g*8.. ; elems 8-15 -> K=16+g*8..
    //  B lane(l): n=l%16; elems 0-7 -> K=g*8.. ; elems 8-15 -> K=16+g*8..
    FragBF a[2], b[2];
    for (int i = 0; i < 2; ++i) {
      int mrow = wm * 32 + i * 16 + mn;
      a[i].h[0] = *(const v8bf*)(&As[p][mrow * GM_LDA + g * 8]);
      a[i].h[1] = *(const v8bf*)(&As[p][mrow * GM_LDA + 16 + g * 8]);
    }
    for (int j = 0; j < 2; ++j) {
      int nrow = wn * 32 + j * 16 + mn;
      b[j].h[0] = *(const v8bf*)(&Bs[p][nrow * GM_LDB + g * 8]);
      b[j].h[1] = *(const v8bf*)(&Bs[p][nrow * GM_LDB + 16 + g * 8]);
    }
    for (int i = 0; i < 2; ++i)
      for (int j = 0; j < 2; ++j)
        acc[i][j] = __builtin_amdgcn_wmma_f32_16x16x32_bf16(
            false, a[i].v, false, b[j].v, (short)0, acc[i][j], false, false);
    __syncthreads();                 // all waves done reading buffer p
    p ^= 1;
  }

  // Epilogue: bias + store; strength-reduced addressing, no branches.
  for (int i = 0; i < 2; ++i) {
    for (int j = 0; j < 2; ++j) {
      const int col  = n0 + wn * 32 + j * 16 + mn;
      const int row0 = m0 + wm * 32 + i * 16 + 8 * g;
      const float bv = bias[col];
      if (HAS_C) {
        float* cp = C + (size_t)row0 * N + col;
#pragma unroll
        for (int jj = 0; jj < 8; ++jj) { *cp = acc[i][j][jj] + bv; cp += N; }
      }
      if (HAS_CB) {
        bf16* cbp = Cb + (size_t)row0 * N + col;
#pragma unroll
        for (int jj = 0; jj < 8; ++jj) { *cbp = (bf16)(acc[i][j][jj] + bv); cbp += N; }
      }
    }
  }
}

// ---------------------------------------------------------------------------
// Recurrence: h_t = XH[:,t,:] + h_{t-1} Whh^T + bhh   (B=16, H=1024)
// Single workgroup, 32 waves; h state transposed in LDS (hT[k][m], 64KB);
// fp32 WMMA 16x16x4; Whh streamed from L2 (4MB, L2-resident).
// ---------------------------------------------------------------------------
__global__ __launch_bounds__(1024) void recurrence_kernel(
    const float* __restrict__ XH,   // [B*T, H], row = b*T + t
    const float* __restrict__ Whh,  // [H, H]
    const float* __restrict__ bhh,  // [H]
    bf16* __restrict__ hs)          // [B*T, H] bf16
{
  __shared__ float hT[cH * 16];     // hT[k*16 + m], exactly 64 KB

  const int tid  = threadIdx.x;
  const int lane = tid & 31;
  const int wave = tid >> 5;        // 0..31, each owns N-tiles 2w, 2w+1
  const int g    = lane >> 4;
  const int mn   = lane & 15;

  for (int i = tid; i < cH * 16; i += 1024) hT[i] = 0.f;
  __syncthreads();

  for (int t = 0; t < cT; ++t) {
    v8f accs[2];
    for (int j = 0; j < 2; ++j) {
      const int nt  = wave * 2 + j;
      const int col = nt * 16 + mn;
      const float bv = bhh[col];
      v8f c;
      for (int jj = 0; jj < 8; ++jj) {
        int b = jj + 8 * g;
        c[jj] = XH[((size_t)b * cT + t) * cH + col] + bv;
      }
      const float* wrow = Whh + (size_t)col * cH;
#pragma unroll 4
      for (int k = 0; k < cH; k += 4) {
        // A (16x4 f32): lane half g -> V0:K=k+2g, V1:K=k+2g+1, row m=mn
        v2f a, b2;
        a.x = hT[(k + 2 * g)     * 16 + mn];
        a.y = hT[(k + 2 * g + 1) * 16 + mn];
        // B (4x16 f32): B[k][n] = Whh[n][k], contiguous pair
        b2.x = wrow[k + 2 * g];
        b2.y = wrow[k + 2 * g + 1];
        c = __builtin_amdgcn_wmma_f32_16x16x4_f32(
            false, a, false, b2, (short)0, c, false, false);
      }
      accs[j] = c;
    }
    __syncthreads();   // everyone done reading h_{t-1}
    for (int j = 0; j < 2; ++j) {
      const int nt  = wave * 2 + j;
      const int col = nt * 16 + mn;
      for (int jj = 0; jj < 8; ++jj) {
        int b = jj + 8 * g;
        hT[col * 16 + b] = accs[j][jj];
        hs[((size_t)b * cT + t) * cH + col] = (bf16)accs[j][jj];
      }
    }
    __syncthreads();   // h_t fully published
  }
}

// ---------------------------------------------------------------------------
// Host launcher
// ---------------------------------------------------------------------------
extern "C" void kernel_launch(void* const* d_in, const int* in_sizes, int n_in,
                              void* d_out, int out_size, void* d_ws, size_t ws_size,
                              hipStream_t stream) {
  const int*   X   = (const int*)  d_in[0];
  const float* emb = (const float*)d_in[1];
  const float* Wxh = (const float*)d_in[2];
  const float* bxh = (const float*)d_in[3];
  const float* Whh = (const float*)d_in[4];
  const float* bhh = (const float*)d_in[5];
  const float* Wyh = (const float*)d_in[6];
  const float* byh = (const float*)d_in[7];
  const float* Wlo = (const float*)d_in[8];
  const float* blo = (const float*)d_in[9];
  float* logits = (float*)d_out;   // [B*T, V] f32

  // Workspace carve-out (256B aligned)
  size_t off = 0;
  auto wsalloc = [&](size_t bytes) -> char* {
    char* p = (char*)d_ws + off;
    off += (bytes + 255) & ~(size_t)255;
    return p;
  };
  const size_t BT = (size_t)cB * cT;          // 8192
  bf16*  Xe   = (bf16*) wsalloc(BT * cE * 2);          //   8 MB
  bf16*  Wxhb = (bf16*) wsalloc((size_t)cH * cE * 2);  //   1 MB
  bf16*  Wyhb = (bf16*) wsalloc((size_t)cV * cH * 2);  //  16 MB
  bf16*  Wlob = (bf16*) wsalloc((size_t)cV * cV * 2);  // 128 MB
  float* XH   = (float*)wsalloc(BT * cH * 4);          //  32 MB
  bf16*  hsb  = (bf16*) wsalloc(BT * cH * 2);          //  16 MB
  bf16*  outb = (bf16*) wsalloc(BT * cV * 2);          // 128 MB
  (void)ws_size; (void)in_sizes; (void)n_in; (void)out_size;

  // 1) Weight casts to bf16
  {
    int n1 = cH * cE;
    cast_f32_bf16_kernel<<<(n1 + 255) / 256, 256, 0, stream>>>(Wxh, Wxhb, n1);
    int n2 = cV * cH;
    cast_f32_bf16_kernel<<<(n2 + 255) / 256, 256, 0, stream>>>(Wyh, Wyhb, n2);
    int n3 = cV * cV;
    cast_f32_bf16_kernel<<<(n3 + 255) / 256, 256, 0, stream>>>(Wlo, Wlob, n3);
  }

  // 2) Embedding gather (+cast)
  {
    int n = (int)(BT * cE);
    embed_gather_kernel<<<(n + 255) / 256, 256, 0, stream>>>(X, emb, Xe, n, cE);
  }

  // 3) XH = Xe @ Wxh^T + bxh   [8192,1024] f32
  gemm_bf16_kernel<true, false><<<dim3(cH / GM_BN, (int)BT / GM_BM), 256, 0, stream>>>(
      Xe, Wxhb, bxh, XH, nullptr, (int)BT, cH, cE);

  // 4) Sequential recurrence -> hs (bf16)
  recurrence_kernel<<<1, 1024, 0, stream>>>(XH, Whh, bhh, hsb);

  // 5) out = hs @ Wyh^T + byh   -> bf16 only (feeds final GEMM)
  gemm_bf16_kernel<false, true><<<dim3(cV / GM_BN, (int)BT / GM_BM), 256, 0, stream>>>(
      hsb, Wyhb, byh, nullptr, outb, (int)BT, cV, cH);

  // 6) logits = out @ Wlo^T + blo  -> d_out f32
  gemm_bf16_kernel<true, false><<<dim3(cV / GM_BN, (int)BT / GM_BM), 256, 0, stream>>>(
      outb, Wlob, blo, logits, nullptr, (int)BT, cV, cV);
}
